// GPT2Attention_20504173871391
// MI455X (gfx1250) — matrix operand order repaired and checked by reading, not compiled, not run on hardware
//
#include <hip/hip_runtime.h>
#include <hip/hip_bf16.h>

typedef _Float16 f16;
typedef __attribute__((ext_vector_type(16))) _Float16 v16h;
typedef __attribute__((ext_vector_type(8)))  _Float16 v8h;
typedef __attribute__((ext_vector_type(8)))  float    v8f;

#define WMMA(a, b, c) __builtin_amdgcn_wmma_f32_16x16x32_f16( \
    false, (a), false, (b), (short)0, (c), false, false)

__device__ inline v16h make16(v8h lo, v8h hi) {
  v16h r;
#pragma unroll
  for (int j = 0; j < 8; ++j) { r[j] = lo[j]; r[j + 8] = hi[j]; }
  return r;
}

// ---------------------------------------------------------------------------
// f32 -> f16 conversion (activations)
// ---------------------------------------------------------------------------
__global__ void cvt_f32_f16(const float* __restrict__ s, f16* __restrict__ d, int n) {
  int i = blockIdx.x * blockDim.x + threadIdx.x;
  int stride = gridDim.x * blockDim.x;
  for (; i < n; i += stride) d[i] = (f16)s[i];
}

// ---------------------------------------------------------------------------
// Weights: f32 [K,N] -> f16 transposed [N,K] via 64x64 LDS tile.
// Grid: (K/64, N/64), block 256.
// ---------------------------------------------------------------------------
__global__ __launch_bounds__(256) void cvt_transpose(const float* __restrict__ W,
                                                     f16* __restrict__ Wt,
                                                     int K, int N) {
  __shared__ f16 lw[64 * 72];
  const int t = threadIdx.x;
  const int k0 = blockIdx.x * 64;
  const int n0 = blockIdx.y * 64;
#pragma unroll
  for (int i = 0; i < 2; ++i) {
    int seg = i * 256 + t;               // 512 segments of 8 elems
    int r = seg >> 3;                    // k row 0..63
    int c = (seg & 7) * 8;               // n col offset
    const float* p = W + (size_t)(k0 + r) * N + n0 + c;
    float4 a = *(const float4*)(p);
    float4 b = *(const float4*)(p + 4);
    v8h v;
    v[0] = (f16)a.x; v[1] = (f16)a.y; v[2] = (f16)a.z; v[3] = (f16)a.w;
    v[4] = (f16)b.x; v[5] = (f16)b.y; v[6] = (f16)b.z; v[7] = (f16)b.w;
    *(v8h*)(lw + r * 72 + c) = v;
  }
  __syncthreads();
#pragma unroll
  for (int i = 0; i < 2; ++i) {
    int seg = i * 256 + t;
    int d = seg >> 3;                    // n 0..63
    int s8 = (seg & 7) * 8;              // k offset
    v8h v;
#pragma unroll
    for (int j = 0; j < 8; ++j) v[j] = lw[(s8 + j) * 72 + d];
    *(v8h*)(Wt + (size_t)(n0 + d) * K + k0 + s8) = v;
  }
}

// ---------------------------------------------------------------------------
// f16 GEMM: Out[M,N] = A[M,K] * Bt[N,K]^T + bias[N]
// Block: 256 threads (8 waves). Block tile 128x128, k-step 32,
// double-buffered LDS with register-staged prefetch.
// Wave (wm, wn) owns 64x32: 4 M-subtiles x 2 N-subtiles of 16x16.
// ---------------------------------------------------------------------------
template <bool OUT_F32>
__global__ __launch_bounds__(256) void gemm_f16(
    const f16* __restrict__ A, const f16* __restrict__ Bt,
    const float* __restrict__ bias, void* __restrict__ Out,
    int M, int N, int K) {
  __shared__ f16 lA[2][128 * 40];   // [row][k], stride 40
  __shared__ f16 lB[2][128 * 40];   // [col][k], stride 40

  const int t = threadIdx.x;
  const int lane = t & 31, wid = t >> 5;
  const int l16 = lane & 15, hf = lane >> 4;
  const int wm = wid & 1, wn = wid >> 1;
  const int m0 = blockIdx.y * 128, n0 = blockIdx.x * 128;

  // staging coordinates: thread t covers rows r0 and r0+64, cols c0..c0+7
  const int r0 = t >> 2;
  const int c0 = (t & 3) * 8;

  v8f acc[4][2];
#pragma unroll
  for (int mt = 0; mt < 4; ++mt)
#pragma unroll
    for (int nt = 0; nt < 2; ++nt)
#pragma unroll
      for (int i = 0; i < 8; ++i) acc[mt][nt][i] = 0.0f;

  const int nsteps = K >> 5;
  v8h ra0, ra1, rb0, rb1;

  // prefetch k-step 0 and stage into buffer 0
  ra0 = *(const v8h*)(A + (size_t)(m0 + r0) * K + c0);
  ra1 = *(const v8h*)(A + (size_t)(m0 + r0 + 64) * K + c0);
  rb0 = *(const v8h*)(Bt + (size_t)(n0 + r0) * K + c0);
  rb1 = *(const v8h*)(Bt + (size_t)(n0 + r0 + 64) * K + c0);
  *(v8h*)(&lA[0][r0 * 40 + c0]) = ra0;
  *(v8h*)(&lA[0][(r0 + 64) * 40 + c0]) = ra1;
  *(v8h*)(&lB[0][r0 * 40 + c0]) = rb0;
  *(v8h*)(&lB[0][(r0 + 64) * 40 + c0]) = rb1;

  for (int s = 0; s < nsteps; ++s) {
    const int buf = s & 1;
    const bool more = (s + 1) < nsteps;
    if (more) {
      const int kk = (s + 1) << 5;
      ra0 = *(const v8h*)(A + (size_t)(m0 + r0) * K + kk + c0);
      ra1 = *(const v8h*)(A + (size_t)(m0 + r0 + 64) * K + kk + c0);
      rb0 = *(const v8h*)(Bt + (size_t)(n0 + r0) * K + kk + c0);
      rb1 = *(const v8h*)(Bt + (size_t)(n0 + r0 + 64) * K + kk + c0);
    }
    __syncthreads();   // buffer `buf` fully staged (stores from prior iter)

    // A fragments: lane<16 row m, K {0..7,16..23}; lane>=16: {8..15,24..31}
    v16h afr[4];
#pragma unroll
    for (int mt = 0; mt < 4; ++mt) {
      const f16* p = &lA[buf][(wm * 64 + mt * 16 + l16) * 40 + hf * 8];
      v8h lo = *(const v8h*)(p);
      v8h hi = *(const v8h*)(p + 16);
      afr[mt] = make16(lo, hi);
    }
    // B fragments: lane holds col = base + l16, K {hf*8..+7, 16+hf*8..+7}
    v16h bfr[2];
#pragma unroll
    for (int nt = 0; nt < 2; ++nt) {
      const f16* p = &lB[buf][(wn * 32 + nt * 16 + l16) * 40 + hf * 8];
      v8h lo = *(const v8h*)(p);
      v8h hi = *(const v8h*)(p + 16);
      bfr[nt] = make16(lo, hi);
    }
#pragma unroll
    for (int mt = 0; mt < 4; ++mt)
#pragma unroll
      for (int nt = 0; nt < 2; ++nt)
        acc[mt][nt] = WMMA(afr[mt], bfr[nt], acc[mt][nt]);

    if (more) {       // stage next tile into the other buffer (no barrier:
      const int nb = buf ^ 1;  // readers of `nb` finished before the sync above)
      *(v8h*)(&lA[nb][r0 * 40 + c0]) = ra0;
      *(v8h*)(&lA[nb][(r0 + 64) * 40 + c0]) = ra1;
      *(v8h*)(&lB[nb][r0 * 40 + c0]) = rb0;
      *(v8h*)(&lB[nb][(r0 + 64) * 40 + c0]) = rb1;
    }
  }

  // Epilogue: C layout -> VGPR i holds row (i + hf*8), col = l16
#pragma unroll
  for (int mt = 0; mt < 4; ++mt)
#pragma unroll
    for (int nt = 0; nt < 2; ++nt)
#pragma unroll
      for (int i = 0; i < 8; ++i) {
        int gr = m0 + wm * 64 + mt * 16 + i + hf * 8;
        int gc = n0 + wn * 32 + nt * 16 + l16;
        float v = acc[mt][nt][i] + bias[gc];
        if (OUT_F32)
          ((float*)Out)[(size_t)gr * N + gc] = v;
        else
          ((f16*)Out)[(size_t)gr * N + gc] = (f16)v;
      }
}

// ---------------------------------------------------------------------------
// Repack V slice of qkv into Vt[bh][hd=64][S=2048] (f16), LDS transpose.
// Grid: (S/64, 48), block 256.
// ---------------------------------------------------------------------------
__global__ __launch_bounds__(256) void repack_vt(const f16* __restrict__ qkv,
                                                 f16* __restrict__ vt) {
  __shared__ f16 lv[64 * 72];
  const int t = threadIdx.x;
  const int s0 = blockIdx.x * 64;
  const int bh = blockIdx.y;
  const int b = bh / 12, h = bh % 12;
#pragma unroll
  for (int i = 0; i < 2; ++i) {
    int seg = i * 256 + t;               // 512 segments of 8
    int r = seg >> 3;                    // token 0..63
    int c = (seg & 7) * 8;               // hd 0..56
    v8h v = *(const v8h*)(qkv + (size_t)(b * 2048 + s0 + r) * 2304 + 1536 + h * 64 + c);
    *(v8h*)(lv + r * 72 + c) = v;
  }
  __syncthreads();
#pragma unroll
  for (int i = 0; i < 2; ++i) {
    int seg = i * 256 + t;
    int d = seg >> 3;                    // hd 0..63
    int s8 = (seg & 7) * 8;              // token offset
    v8h v;
#pragma unroll
    for (int j = 0; j < 8; ++j) v[j] = lv[(s8 + j) * 72 + d];
    *(v8h*)(vt + ((size_t)bh * 64 + d) * 2048 + s0 + s8) = v;
  }
}

// ---------------------------------------------------------------------------
// Flash attention: one wave per (query block of 16, b, h).
// Streams 32-key tiles with online softmax; all matmuls via WMMA f16.
// ---------------------------------------------------------------------------
__global__ __launch_bounds__(32) void attn_fa(const f16* __restrict__ qkv,
                                              const f16* __restrict__ vt,
                                              f16* __restrict__ attn) {
  __shared__ f16 lP[16 * 40];            // probabilities tile 16 x 32
  const int lane = threadIdx.x & 31;
  const int l16 = lane & 15, hf = lane >> 4;
  const int qb = blockIdx.x;             // 0..127
  const int bh = blockIdx.y;             // 0..47
  const int b = bh / 12, h = bh % 12;

  // Q A-fragments for hd k-steps 0..31 and 32..63
  const int qrow = qb * 16 + l16;
  const f16* qptr = qkv + (size_t)(b * 2048 + qrow) * 2304 + h * 64;
  v16h aq[2];
#pragma unroll
  for (int ks = 0; ks < 2; ++ks) {
    v8h lo = *(const v8h*)(qptr + ks * 32 + hf * 8);
    v8h hi = *(const v8h*)(qptr + ks * 32 + 16 + hf * 8);
    aq[ks] = make16(lo, hi);
  }

  float mmax[8], lsum[8];
  v8f acc[4];
#pragma unroll
  for (int i = 0; i < 8; ++i) { mmax[i] = -3.0e38f; lsum[i] = 0.0f; }
#pragma unroll
  for (int nt = 0; nt < 4; ++nt)
#pragma unroll
    for (int i = 0; i < 8; ++i) acc[nt][i] = 0.0f;

  const int nj = qb / 2 + 1;             // causal frontier in 32-key tiles
  for (int j = 0; j < nj; ++j) {
    const int kb = j * 32;

    // scores: two 16-key subtiles, each QK^T over hd=64 (2 WMMAs)
    v8f c[2];
#pragma unroll
    for (int ns = 0; ns < 2; ++ns) {
      const int key = kb + ns * 16 + l16;
      const f16* kp = qkv + (size_t)(b * 2048 + key) * 2304 + 768 + h * 64;
      v8f z;
#pragma unroll
      for (int i = 0; i < 8; ++i) z[i] = 0.0f;
#pragma unroll
      for (int ks = 0; ks < 2; ++ks) {
        v8h lo = *(const v8h*)(kp + ks * 32 + hf * 8);
        v8h hi = *(const v8h*)(kp + ks * 32 + 16 + hf * 8);
        z = WMMA(aq[ks], make16(lo, hi), z);
      }
      c[ns] = z;
    }

    // online softmax (row i+hf*8 lives across one 16-lane half)
#pragma unroll
    for (int i = 0; i < 8; ++i) {
      const int qi = qb * 16 + i + hf * 8;
      float s0 = c[0][i] * 0.125f;
      float s1 = c[1][i] * 0.125f;
      if (kb + l16 > qi)      s0 = -3.0e38f;
      if (kb + 16 + l16 > qi) s1 = -3.0e38f;
      float tm = fmaxf(s0, s1);
      tm = fmaxf(tm, __shfl_xor(tm, 1, 32));
      tm = fmaxf(tm, __shfl_xor(tm, 2, 32));
      tm = fmaxf(tm, __shfl_xor(tm, 4, 32));
      tm = fmaxf(tm, __shfl_xor(tm, 8, 32));
      float nm = fmaxf(mmax[i], tm);
      float al = __expf(mmax[i] - nm);
      float p0 = __expf(s0 - nm), p1 = __expf(s1 - nm);
      float rs = p0 + p1;
      rs += __shfl_xor(rs, 1, 32);
      rs += __shfl_xor(rs, 2, 32);
      rs += __shfl_xor(rs, 4, 32);
      rs += __shfl_xor(rs, 8, 32);
      lsum[i] = lsum[i] * al + rs;
      mmax[i] = nm;
#pragma unroll
      for (int nt = 0; nt < 4; ++nt) acc[nt][i] *= al;
      const int m = i + hf * 8;          // C-layout -> LDS (f16)
      lP[m * 40 + l16] = (f16)p0;
      lP[m * 40 + 16 + l16] = (f16)p1;
    }

    // reload P as an A fragment (16x32 over keys)
    v8h plo = *(const v8h*)(lP + l16 * 40 + hf * 8);
    v8h phi = *(const v8h*)(lP + l16 * 40 + 16 + hf * 8);
    v16h ap = make16(plo, phi);

    // PV: B fragments are key-runs {hf*8..+7, 16+hf*8..+7} of V^T
#pragma unroll
    for (int nt = 0; nt < 4; ++nt) {
      const f16* vp = vt + ((size_t)bh * 64 + nt * 16 + l16) * 2048 + kb + hf * 8;
      v8h blo = *(const v8h*)(vp);
      v8h bhi = *(const v8h*)(vp + 16);
      acc[nt] = WMMA(ap, make16(blo, bhi), acc[nt]);
    }
  }

  // normalize and write merged-head output [8192 x 768] f16
#pragma unroll
  for (int i = 0; i < 8; ++i) {
    const float rl = 1.0f / lsum[i];
    const int q = qb * 16 + i + hf * 8;
#pragma unroll
    for (int nt = 0; nt < 4; ++nt) {
      const int d = h * 64 + nt * 16 + l16;
      attn[(size_t)(b * 2048 + q) * 768 + d] = (f16)(acc[nt][i] * rl);
    }
  }
}

// ---------------------------------------------------------------------------
// Host launch
// ---------------------------------------------------------------------------
extern "C" void kernel_launch(void* const* d_in, const int* in_sizes, int n_in,
                              void* d_out, int out_size, void* d_ws, size_t ws_size,
                              hipStream_t stream) {
  (void)in_sizes; (void)n_in; (void)out_size; (void)ws_size;
  const float* X  = (const float*)d_in[0];   // [4,2048,768]
  const float* Wq = (const float*)d_in[1];   // [768,2304]
  const float* bq = (const float*)d_in[2];   // [2304]
  const float* Wp = (const float*)d_in[3];   // [768,768]
  const float* bp = (const float*)d_in[4];   // [768]
  float* out = (float*)d_out;                // [4,2048,768] f32

  char* ws = (char*)d_ws;
  f16* Xh   = (f16*)(ws);                    // 12,582,912 B  [8192 x 768]
  f16* Wqt  = (f16*)(ws + 12582912);         //  3,538,944 B  [2304 x 768] (W^T)
  f16* Wpt  = (f16*)(ws + 16121856);         //  1,179,648 B  [768 x 768]  (W^T)
  f16* qkv  = (f16*)(ws + 17301504);         // 37,748,736 B  [8192 x 2304]
  f16* vt   = (f16*)(ws + 55050240);         // 12,582,912 B  [48 x 64 x 2048]
  f16* attn = (f16*)(ws + 67633152);         // 12,582,912 B  [8192 x 768]

  cvt_f32_f16<<<2048, 256, 0, stream>>>(X, Xh, 8192 * 768);
  cvt_transpose<<<dim3(12, 36), 256, 0, stream>>>(Wq, Wqt, 768, 2304);
  cvt_transpose<<<dim3(12, 12), 256, 0, stream>>>(Wp, Wpt, 768, 768);

  // QKV projection: [8192,768] x [768,2304] + bias -> f16 qkv
  gemm_f16<false><<<dim3(18, 64), 256, 0, stream>>>(Xh, Wqt, bq, (void*)qkv,
                                                    8192, 2304, 768);
  // V^T repack for PV matmul
  repack_vt<<<dim3(32, 48), 256, 0, stream>>>(qkv, vt);
  // Flash attention
  attn_fa<<<dim3(128, 48), 32, 0, stream>>>(qkv, vt, attn);
  // Output projection: [8192,768] x [768,768] + bias -> f32 out
  gemm_f16<true><<<dim3(6, 64), 256, 0, stream>>>(attn, Wpt, bp, (void*)out,
                                                  8192, 768, 768);
}
